// GPTModel_2534030705251
// MI455X (gfx1250) — matrix operand order repaired and checked by reading, not compiled, not run on hardware
//
#include <hip/hip_runtime.h>
#include <hip/hip_bf16.h>
#include <math.h>

// ---------------------------------------------------------------------------
// GPT-2-ish forward for MI455X (gfx1250), wave32 + WMMA bf16 + async-to-LDS.
//   L=6, D=768, H=12, HD=64, V=50257, T=1024, B=2  ->  M = B*T = 2048 rows.
// Pipeline per call:
//   1. per-weight f32 -> bf16 transpose ([K][N] -> [N][K]) into a reused
//      staging buffer (weights are L2-resident and re-read 16x by M-blocks,
//      so converting once halves L2 traffic and cleans the GEMM hot loop)
//   2. all GEMMs: 128x128 block tile, 8 waves, double-buffered LDS staged via
//      global_load_async_to_lds_b128 (ASYNCcnt), v_wmma_f32_16x16x32_bf16
//   3. flash attention: 1 wave / (b, head, 16 queries), online softmax,
//      V consumed from a transposed buffer so all fragments are b128 loads
// Workspace requirement: ~112 MB (see layout in kernel_launch).
// ---------------------------------------------------------------------------

typedef __attribute__((ext_vector_type(16))) __bf16 v16bf;
typedef __attribute__((ext_vector_type(8)))  float  v8f;

#define GPT_L  6
#define GPT_D  768
#define GPT_H  12
#define GPT_HD 64
#define GPT_V  50257
#define GPT_T  1024
#define GPT_B  2
#define GPT_M  (GPT_B * GPT_T)   // 2048 token rows
#define GPT_F  (4 * GPT_D)       // 3072

__device__ __forceinline__ v8f wmma_bf16(v16bf a, v16bf b, v8f c) {
  // D = A(16x32) * B(32x16) + C, f32 accumulate
  return __builtin_amdgcn_wmma_f32_16x16x32_bf16(
      /*neg_a=*/false, a, /*neg_b=*/false, b,
      /*c_mod=*/(short)0, c, /*reuse_a=*/false, /*reuse_b=*/false);
}

// A-fragment (16x32, 16-bit): lane&15 = row M; holds K {kb..kb+7, kb+16..kb+23},
// kb = 8 for lanes 16-31.  Two 16-byte chunks at p and p+16 elements.
__device__ __forceinline__ v16bf frag_ld_a(const __bf16* p) {
  union { uint4 u[2]; v16bf v; } t;
  t.u[0] = *(const uint4*)(p);
  t.u[1] = *(const uint4*)(p + 16);
  return t.v;
}

// B-fragment (32x16, 16-bit): lane&15 = col N; holds K {kb..kb+7, kb+16..kb+23},
// kb = 8 for lanes 16-31.  Two 16-byte chunks at p and p+16 elements.
__device__ __forceinline__ v16bf frag_ld_b(const __bf16* p) {
  union { uint4 u[2]; v16bf v; } t;
  t.u[0] = *(const uint4*)(p);
  t.u[1] = *(const uint4*)(p + 16);
  return t.v;
}

// CDNA5 async DMA: copy 16 bytes global -> LDS, tracked by ASYNCcnt.
__device__ __forceinline__ void async_b128(unsigned lds_off, const void* gsrc) {
  asm volatile("global_load_async_to_lds_b128 %0, %1, off"
               :: "v"(lds_off), "v"(gsrc)
               : "memory");
}
__device__ __forceinline__ void wait_async0() {
  asm volatile("s_wait_asynccnt 0" ::: "memory");
}

__device__ __forceinline__ float gelu_f(float x) {
  float x3 = x * x * x;
  return 0.5f * x * (1.0f + tanhf(0.7978845608028654f * (x + 0.044715f * x3)));
}

// ---------------------------------------------------------------------------
// Embedding gather: x[row,:] = tok_emb[idx[row],:] + pos_emb[row%T,:]
// ---------------------------------------------------------------------------
__global__ void embed_kernel(const int* __restrict__ idx,
                             const float* __restrict__ tok,
                             const float* __restrict__ pos,
                             float* __restrict__ x) {
  const int row = blockIdx.x;
  const int t   = row & (GPT_T - 1);
  const int tid = idx[row];
  for (int i = threadIdx.x; i < GPT_D; i += 256)
    x[(size_t)row * GPT_D + i] =
        tok[(size_t)tid * GPT_D + i] + pos[(size_t)t * GPT_D + i];
}

// ---------------------------------------------------------------------------
// LayerNorm over D=768, one workgroup (8 waves) per row, emits bf16.
// ---------------------------------------------------------------------------
__global__ void ln_kernel(const float* __restrict__ x,
                          const float* __restrict__ sc,
                          const float* __restrict__ sh,
                          __bf16* __restrict__ out) {
  const int row = blockIdx.x;
  const float* xr = x + (size_t)row * GPT_D;
  __shared__ float red[256];
  float sum = 0.f, sq = 0.f;
  for (int i = threadIdx.x; i < GPT_D; i += 256) {
    float v = xr[i];
    sum += v;
    sq  += v * v;
  }
  red[threadIdx.x] = sum;
  __syncthreads();
  for (int o = 128; o > 0; o >>= 1) {
    if (threadIdx.x < o) red[threadIdx.x] += red[threadIdx.x + o];
    __syncthreads();
  }
  const float mean = red[0] * (1.0f / GPT_D);
  __syncthreads();
  red[threadIdx.x] = sq;
  __syncthreads();
  for (int o = 128; o > 0; o >>= 1) {
    if (threadIdx.x < o) red[threadIdx.x] += red[threadIdx.x + o];
    __syncthreads();
  }
  const float var = red[0] * (1.0f / GPT_D) - mean * mean;  // biased
  const float inv = rsqrtf(var + 1e-5f);
  for (int i = threadIdx.x; i < GPT_D; i += 256)
    out[(size_t)row * GPT_D + i] =
        (__bf16)(sc[i] * (xr[i] - mean) * inv + sh[i]);
}

// ---------------------------------------------------------------------------
// Weight prep: W[K][N] f32  ->  Wt[N][K] bf16 (32x32 LDS tile transpose).
// Runs once per weight per call; K is always a multiple of 32, N may be ragged.
// ---------------------------------------------------------------------------
__global__ void __launch_bounds__(256)
transpose_cvt_kernel(const float* __restrict__ W, __bf16* __restrict__ Wt,
                     int K, int N) {
  __shared__ float t[32][33];
  const int nb = blockIdx.x * 32;
  const int kb = blockIdx.y * 32;
  const int tx = threadIdx.x;          // 0..31
  const int ty = threadIdx.y;          // 0..7
  for (int r = ty; r < 32; r += 8) {
    int n = nb + tx;
    t[r][tx] = (n < N) ? W[(size_t)(kb + r) * N + n] : 0.f;
  }
  __syncthreads();
  for (int r = ty; r < 32; r += 8) {
    int n = nb + r;
    if (n < N) Wt[(size_t)n * K + kb + tx] = (__bf16)t[tx][r];
  }
}

// ---------------------------------------------------------------------------
// WMMA GEMM:  C[M,N] = epi(A[M,K](bf16) @ Wt[N,K](bf16,row=N) [+bias][gelu][+res])
// Block tile 128x128, 8 waves (4x2), wave tile 32x64 = 8 wmma accumulators.
// Double-buffered LDS; tiles staged with global_load_async_to_lds_b128.
// OUT_MODE: 0 = f32 [row*N+col], 1 = bf16 [row*N+col], 2 = bf16 [col*M+row].
// M, K multiples of 128/32; N may be ragged (head: 50257).
// ---------------------------------------------------------------------------
template <bool HAS_BIAS, bool DO_GELU, bool RESID, int OUT_MODE>
__global__ void __launch_bounds__(256)
gemm_kernel(const __bf16* __restrict__ A, const __bf16* __restrict__ Wt,
            const float* __restrict__ bias, const float* __restrict__ resid,
            void* __restrict__ Cout, int M, int N, int K) {
  constexpr int LDSS = 40;              // padded K stride (80 B, bank-spread)
  constexpr int TILE = 128 * LDSS;      // elements per buffer
  __shared__ __bf16 ldsA[2 * TILE];
  __shared__ __bf16 ldsB[2 * TILE];

  const int tid  = threadIdx.x;
  const int lane = tid & 31;
  const int w    = tid >> 5;
  const int wm   = w >> 1;        // 0..3 -> 32-row slice
  const int wn   = w & 1;         // 0..1 -> 64-col slice
  const int half = (lane >> 4) & 1;
  const int aoff = half * 8;      // A-frag K base
  const int boff = half * 8;      // B-frag K base
  const int n15  = lane & 15;
  const int m0   = blockIdx.y * 128;
  const int n0   = blockIdx.x * 128;

  const unsigned ldsA0 = (unsigned)(uintptr_t)(&ldsA[0]);
  const unsigned ldsB0 = (unsigned)(uintptr_t)(&ldsB[0]);

  // Stage one 128x32 bf16 tile of A and Wt into LDS buffer `buf`.
  // 512 16-byte chunks each; 2 chunks/thread/matrix.
  auto stage = [&](int k0, int buf) {
#pragma unroll
    for (int j = 0; j < 2; ++j) {
      int cid = tid + j * 256;
      int row = cid >> 2;
      int ck  = (cid & 3) * 8;
      unsigned loff = (unsigned)((buf * TILE + row * LDSS + ck) * 2);
      async_b128(ldsA0 + loff, A + (size_t)(m0 + row) * K + k0 + ck);
      int rb = n0 + row;
      if (rb >= N) rb = N - 1;  // clamp: junk cols are masked at store
      async_b128(ldsB0 + loff, Wt + (size_t)rb * K + k0 + ck);
    }
  };

  v8f acc[2][4];
  v8f zero = {};
#pragma unroll
  for (int s = 0; s < 2; ++s)
#pragma unroll
    for (int f = 0; f < 4; ++f) acc[s][f] = zero;

  stage(0, 0);
  int cur = 0;
  for (int k0 = 0; k0 < K; k0 += 32) {
    wait_async0();        // our async writes for buf `cur` are in LDS
    __syncthreads();      // everyone's are; previous compute done
    if (k0 + 32 < K) stage(k0 + 32, cur ^ 1);

    const __bf16* bufA = ldsA + cur * TILE;
    const __bf16* bufB = ldsB + cur * TILE;
    v16bf bfrag[4];
#pragma unroll
    for (int f = 0; f < 4; ++f)
      bfrag[f] = frag_ld_b(&bufB[(wn * 64 + f * 16 + n15) * LDSS + boff]);
#pragma unroll
    for (int s = 0; s < 2; ++s) {
      v16bf afrag = frag_ld_a(&bufA[(wm * 32 + s * 16 + n15) * LDSS + aoff]);
#pragma unroll
      for (int f = 0; f < 4; ++f)
        acc[s][f] = wmma_bf16(afrag, bfrag[f], acc[s][f]);
    }
    cur ^= 1;
  }

  // --- epilogue: C layout is M = i + 8*half (per VGPR i), N = lane&15 ---
#pragma unroll
  for (int s = 0; s < 2; ++s)
#pragma unroll
    for (int f = 0; f < 4; ++f)
#pragma unroll
      for (int i = 0; i < 8; ++i) {
        int row = m0 + wm * 32 + s * 16 + i + half * 8;
        int col = n0 + wn * 64 + f * 16 + n15;
        if (col < N) {
          float v = acc[s][f][i];
          if (HAS_BIAS) v += bias[col];
          if (DO_GELU) v = gelu_f(v);
          if (RESID) v += resid[(size_t)row * N + col];
          if (OUT_MODE == 0)
            ((float*)Cout)[(size_t)row * N + col] = v;
          else if (OUT_MODE == 1)
            ((__bf16*)Cout)[(size_t)row * N + col] = (__bf16)v;
          else  // transposed bf16 (per-lane stores contiguous along rows)
            ((__bf16*)Cout)[(size_t)col * M + row] = (__bf16)v;
        }
      }
}

// ---------------------------------------------------------------------------
// Causal flash attention. One wave per (batch, head, 16-query tile).
// S = Q(16x64) @ K^T via 4 wmma/key-tile; online softmax; P goes through a
// 1 KB LDS buffer to convert C-layout -> A-layout; O += P(16x32) @ V(32x64)
// via 4 wmma/key-tile with V read from the transposed buffer Vt[d][token].
// scale = 1/sqrt(64) = 0.125.
// ---------------------------------------------------------------------------
__global__ void __launch_bounds__(32)
attn_kernel(const __bf16* __restrict__ Qm, const __bf16* __restrict__ Km,
            const __bf16* __restrict__ Vt, __bf16* __restrict__ Cm) {
  const int lane = threadIdx.x & 31;
  const int half = (lane >> 4) & 1;
  const int aoff = half * 8;   // A-frag K base
  const int boff = half * 8;   // B-frag K base
  const int n15  = lane & 15;
  const int qtile   = blockIdx.x;        // 0..63
  const int hoff    = blockIdx.y * GPT_HD;
  const int tokbase = blockIdx.z * GPT_T;

  __shared__ __bf16 ldsP[16 * 32];

  // Q fragments (row = query, K = head dim; two 32-wide K steps)
  const int qrow = tokbase + qtile * 16 + n15;
  const v16bf qa0 = frag_ld_a(Qm + (size_t)qrow * GPT_D + hoff + aoff);
  const v16bf qa1 = frag_ld_a(Qm + (size_t)qrow * GPT_D + hoff + 32 + aoff);

  float mrow[8], lrow[8];
  v8f o[4];
  v8f zero = {};
#pragma unroll
  for (int i = 0; i < 8; ++i) { mrow[i] = -INFINITY; lrow[i] = 0.f; }
#pragma unroll
  for (int n = 0; n < 4; ++n) o[n] = zero;

  const int ktmax = (qtile * 16 + 15) >> 5;
  const float scale = 0.125f;  // 1/sqrt(HD)

  for (int kt = 0; kt <= ktmax; ++kt) {
    // ---- scores for 32 keys ----
    v8f sa[2] = {zero, zero};
#pragma unroll
    for (int n = 0; n < 2; ++n) {
      int ktok = tokbase + kt * 32 + n * 16 + n15;  // lane's key (B-matrix col)
      v16bf kb0 = frag_ld_b(Km + (size_t)ktok * GPT_D + hoff + boff);
      v16bf kb1 = frag_ld_b(Km + (size_t)ktok * GPT_D + hoff + 32 + boff);
      sa[n] = wmma_bf16(qa0, kb0, sa[n]);
      sa[n] = wmma_bf16(qa1, kb1, sa[n]);
    }
    // ---- mask + scale; per-row online softmax (rows live on 16-lane halves)
    float p[2][8], rmax[8], rsum[8], alpha[8];
#pragma unroll
    for (int i = 0; i < 8; ++i) rmax[i] = -INFINITY;
#pragma unroll
    for (int n = 0; n < 2; ++n)
#pragma unroll
      for (int i = 0; i < 8; ++i) {
        int r = qtile * 16 + i + half * 8;
        int c = kt * 32 + n * 16 + n15;
        float v = (c <= r) ? sa[n][i] * scale : -INFINITY;
        p[n][i] = v;
        rmax[i] = fmaxf(rmax[i], v);
      }
#pragma unroll
    for (int off = 1; off < 16; off <<= 1)
#pragma unroll
      for (int i = 0; i < 8; ++i)
        rmax[i] = fmaxf(rmax[i], __shfl_xor(rmax[i], off, 32));
#pragma unroll
    for (int i = 0; i < 8; ++i) {
      float mnew = fmaxf(mrow[i], rmax[i]);
      alpha[i] = __expf(mrow[i] - mnew);
      mrow[i] = mnew;
      p[0][i] = __expf(p[0][i] - mnew);
      p[1][i] = __expf(p[1][i] - mnew);
      rsum[i] = p[0][i] + p[1][i];
    }
#pragma unroll
    for (int off = 1; off < 16; off <<= 1)
#pragma unroll
      for (int i = 0; i < 8; ++i)
        rsum[i] += __shfl_xor(rsum[i], off, 32);
#pragma unroll
    for (int i = 0; i < 8; ++i) lrow[i] = lrow[i] * alpha[i] + rsum[i];
#pragma unroll
    for (int n = 0; n < 4; ++n)
#pragma unroll
      for (int i = 0; i < 8; ++i) o[n][i] = o[n][i] * alpha[i];

    // ---- P: C-layout -> LDS -> A-layout fragment ----
#pragma unroll
    for (int n = 0; n < 2; ++n)
#pragma unroll
      for (int i = 0; i < 8; ++i)
        ldsP[(i + half * 8) * 32 + n * 16 + n15] = (__bf16)p[n][i];
    asm volatile("s_wait_dscnt 0" ::: "memory");
    v16bf pa = frag_ld_a(&ldsP[n15 * 32 + aoff]);

    // ---- O += P @ V (V from transposed buffer: contiguous along keys) ----
#pragma unroll
    for (int n = 0; n < 4; ++n) {
      int col = hoff + n * 16 + n15;  // head-dim column
      v16bf vbf = frag_ld_b(Vt + (size_t)col * GPT_M + tokbase + kt * 32 + boff);
      o[n] = wmma_bf16(pa, vbf, o[n]);
    }
  }

  // ---- normalize and store ctx (bf16, [b,t,h*HD+hd] row-major) ----
#pragma unroll
  for (int n = 0; n < 4; ++n)
#pragma unroll
    for (int i = 0; i < 8; ++i) {
      int r = tokbase + qtile * 16 + i + half * 8;
      int c = hoff + n * 16 + n15;
      Cm[(size_t)r * GPT_D + c] = (__bf16)(o[n][i] / lrow[i]);
    }
}

// ---------------------------------------------------------------------------
// Host-side orchestration (graph-capture safe: stream launches only).
// ---------------------------------------------------------------------------
extern "C" void kernel_launch(void* const* d_in, const int* in_sizes, int n_in,
                              void* d_out, int out_size, void* d_ws,
                              size_t ws_size, hipStream_t stream) {
  (void)in_sizes; (void)n_in; (void)out_size; (void)ws_size;

  const int*   in_idx  = (const int*)d_in[0];
  const float* tok_emb = (const float*)d_in[1];
  const float* pos_emb = (const float*)d_in[2];
  const float* Wq      = (const float*)d_in[3];
  const float* Wk      = (const float*)d_in[4];
  const float* Wv      = (const float*)d_in[5];
  const float* Wo      = (const float*)d_in[6];
  const float* bo      = (const float*)d_in[7];
  const float* W1      = (const float*)d_in[8];
  const float* b1      = (const float*)d_in[9];
  const float* W2      = (const float*)d_in[10];
  const float* b2      = (const float*)d_in[11];
  const float* ln1_s   = (const float*)d_in[12];
  const float* ln1_b   = (const float*)d_in[13];
  const float* ln2_s   = (const float*)d_in[14];
  const float* ln2_b   = (const float*)d_in[15];
  const float* fn_s    = (const float*)d_in[16];
  const float* fn_b    = (const float*)d_in[17];
  const float* W_head  = (const float*)d_in[18];

  // Workspace layout (total ~112 MB, all chunks 256B-aligned):
  char* ws = (char*)d_ws;
  size_t off = 0;
  float*  x    = (float*)(ws + off);  off += (size_t)GPT_M * GPT_D * 4;  // 6.3 MB
  __bf16* hln  = (__bf16*)(ws + off); off += (size_t)GPT_M * GPT_D * 2;  // 3.1 MB
  __bf16* h1   = (__bf16*)(ws + off); off += (size_t)GPT_M * GPT_F * 2;  // 12.6 MB
  __bf16* qb   = (__bf16*)(ws + off); off += (size_t)GPT_M * GPT_D * 2;
  __bf16* kbuf = (__bf16*)(ws + off); off += (size_t)GPT_M * GPT_D * 2;
  __bf16* vt   = (__bf16*)(ws + off); off += (size_t)GPT_M * GPT_D * 2;  // [D][M]
  __bf16* ctxb = (__bf16*)(ws + off); off += (size_t)GPT_M * GPT_D * 2;
  // reused weight staging buffer: max(per-layer 14.2 MB, head 77.2 MB)
  __bf16* wst  = (__bf16*)(ws + off); off += (size_t)GPT_V * GPT_D * 2;  // 77.2 MB
  __bf16* wq_t = wst;                               // [768][768]
  __bf16* wk_t = wq_t + (size_t)GPT_D * GPT_D;
  __bf16* wv_t = wk_t + (size_t)GPT_D * GPT_D;
  __bf16* wo_t = wv_t + (size_t)GPT_D * GPT_D;
  __bf16* w1_t = wo_t + (size_t)GPT_D * GPT_D;      // [3072][768]
  __bf16* w2_t = w1_t + (size_t)GPT_D * GPT_F;      // [768][3072]

  const dim3 blk256(256);
  const dim3 blkT(32, 8);
  const dim3 gD(GPT_D / 128, GPT_M / 128);          // 6 x 16
  const dim3 gF(GPT_F / 128, GPT_M / 128);          // 24 x 16
  const dim3 gHd((GPT_V + 127) / 128, GPT_M / 128); // 393 x 16
  const dim3 gAttn(GPT_T / 16, GPT_H, GPT_B);       // 64 x 12 x 2
  const dim3 gT_dd(GPT_D / 32, GPT_D / 32);         // 24 x 24
  const dim3 gT_df(GPT_F / 32, GPT_D / 32);         // W1: K=768, N=3072
  const dim3 gT_fd(GPT_D / 32, GPT_F / 32);         // W2: K=3072, N=768
  const dim3 gT_hd((GPT_V + 31) / 32, GPT_D / 32);  // head: K=768, N=50257

  embed_kernel<<<GPT_M, blk256, 0, stream>>>(in_idx, tok_emb, pos_emb, x);

  for (int l = 0; l < GPT_L; ++l) {
    const size_t wdd = (size_t)l * GPT_D * GPT_D;
    const size_t wdf = (size_t)l * GPT_D * GPT_F;
    // weight prep for this layer (f32 [K][N] -> bf16 [N][K])
    transpose_cvt_kernel<<<gT_dd, blkT, 0, stream>>>(Wq + wdd, wq_t, GPT_D, GPT_D);
    transpose_cvt_kernel<<<gT_dd, blkT, 0, stream>>>(Wk + wdd, wk_t, GPT_D, GPT_D);
    transpose_cvt_kernel<<<gT_dd, blkT, 0, stream>>>(Wv + wdd, wv_t, GPT_D, GPT_D);
    transpose_cvt_kernel<<<gT_dd, blkT, 0, stream>>>(Wo + wdd, wo_t, GPT_D, GPT_D);
    transpose_cvt_kernel<<<gT_df, blkT, 0, stream>>>(W1 + wdf, w1_t, GPT_D, GPT_F);
    transpose_cvt_kernel<<<gT_fd, blkT, 0, stream>>>(W2 + wdf, w2_t, GPT_F, GPT_D);
    // attention sublayer
    ln_kernel<<<GPT_M, blk256, 0, stream>>>(x, ln1_s + l * GPT_D,
                                            ln1_b + l * GPT_D, hln);
    gemm_kernel<false, false, false, 1><<<gD, blk256, 0, stream>>>(
        hln, wq_t, nullptr, nullptr, qb, GPT_M, GPT_D, GPT_D);
    gemm_kernel<false, false, false, 1><<<gD, blk256, 0, stream>>>(
        hln, wk_t, nullptr, nullptr, kbuf, GPT_M, GPT_D, GPT_D);
    gemm_kernel<false, false, false, 2><<<gD, blk256, 0, stream>>>(
        hln, wv_t, nullptr, nullptr, vt, GPT_M, GPT_D, GPT_D);
    attn_kernel<<<gAttn, dim3(32), 0, stream>>>(qb, kbuf, vt, ctxb);
    gemm_kernel<true, false, true, 0><<<gD, blk256, 0, stream>>>(
        ctxb, wo_t, bo + l * GPT_D, x, x, GPT_M, GPT_D, GPT_D);
    // FFN sublayer
    ln_kernel<<<GPT_M, blk256, 0, stream>>>(x, ln2_s + l * GPT_D,
                                            ln2_b + l * GPT_D, hln);
    gemm_kernel<true, true, false, 1><<<gF, blk256, 0, stream>>>(
        hln, w1_t, b1 + l * GPT_F, nullptr, h1, GPT_M, GPT_F, GPT_D);
    gemm_kernel<true, false, true, 0><<<gD, blk256, 0, stream>>>(
        h1, w2_t, b2 + l * GPT_D, x, x, GPT_M, GPT_D, GPT_F);
  }

  ln_kernel<<<GPT_M, blk256, 0, stream>>>(x, fn_s, fn_b, hln);
  transpose_cvt_kernel<<<gT_hd, blkT, 0, stream>>>(W_head, wst, GPT_D, GPT_V);
  gemm_kernel<false, false, false, 0><<<gHd, blk256, 0, stream>>>(
      hln, wst, nullptr, nullptr, (float*)d_out, GPT_M, GPT_V, GPT_D);
}
